// FluxBitBlock_35347580846131
// MI455X (gfx1250) — 8-bit matrix operand order repaired and checked by reading, compiled, not run on hardware
//
#include <hip/hip_runtime.h>
#include <hip/hip_bf16.h>
#include <math.h>

#define D_      1024
#define B_      2
#define SIMG_   4096
#define SCTX_   512
#define SJ_     (SIMG_ + SCTX_)
#define ROWSJ_  (B_ * SJ_)
#define ROWSH_  (B_ * SIMG_)
#define ROWSC_  (B_ * SCTX_)
#define EPS_    1e-6f

typedef __attribute__((ext_vector_type(8))) int v8i_t;
typedef __attribute__((ext_vector_type(4))) int v4i_t;

union V8U { v8i_t v; long long l[4]; v4i_t q[2]; int i[8]; };

__device__ __forceinline__ float sigmoidf_(float x) { return 1.f / (1.f + __expf(-x)); }
__device__ __forceinline__ float siluf_(float x)    { return x * sigmoidf_(x); }

__device__ __forceinline__ float blk_sum256(float v, float* sh) {
  #pragma unroll
  for (int o = 16; o > 0; o >>= 1) v += __shfl_xor(v, o, 32);
  int wv = threadIdx.x >> 5;
  if ((threadIdx.x & 31) == 0) sh[wv] = v;
  __syncthreads();
  if (threadIdx.x == 0) {
    float s = 0.f;
    for (int i = 0; i < 8; ++i) s += sh[i];
    sh[0] = s;
  }
  __syncthreads();
  float r = sh[0];
  __syncthreads();
  return r;
}

__device__ __forceinline__ float blk_max256(float v, float* sh) {
  #pragma unroll
  for (int o = 16; o > 0; o >>= 1) v = fmaxf(v, __shfl_xor(v, o, 32));
  int wv = threadIdx.x >> 5;
  if ((threadIdx.x & 31) == 0) sh[wv] = v;
  __syncthreads();
  if (threadIdx.x == 0) {
    float s = sh[0];
    for (int i = 1; i < 8; ++i) s = fmaxf(s, sh[i]);
    sh[0] = s;
  }
  __syncthreads();
  float r = sh[0];
  __syncthreads();
  return r;
}

__global__ void k_adaln(const float* __restrict__ temb, const float* __restrict__ w,
                        const float* __restrict__ bias, float* __restrict__ e) {
  __shared__ float st[D_];
  int b = blockIdx.y;
  for (int d = threadIdx.x; d < D_; d += blockDim.x) {
    float x = temb[b * D_ + d];
    st[d] = siluf_(x);
  }
  __syncthreads();
  int o = blockIdx.x * blockDim.x + threadIdx.x;
  const float* wr = w + (size_t)o * D_;
  float acc = 0.f;
  #pragma unroll 8
  for (int d = 0; d < D_; ++d) acc = fmaf(st[d], wr[d], acc);
  e[(size_t)b * 6 * D_ + o] = acc + bias[o];
}

__global__ void k_abs_partial(const float* __restrict__ w, int n, float* __restrict__ partial) {
  __shared__ float sh[8];
  float s = 0.f;
  for (int i = blockIdx.x * blockDim.x + threadIdx.x; i < n; i += gridDim.x * blockDim.x)
    s += fabsf(w[i]);
  s = blk_sum256(s, sh);
  if (threadIdx.x == 0) partial[blockIdx.x] = s;
}

__global__ void k_wscale(const float* __restrict__ partial, float invn, float* __restrict__ deq) {
  __shared__ float sh[256];
  sh[threadIdx.x] = partial[threadIdx.x];
  __syncthreads();
  for (int s = 128; s > 0; s >>= 1) {
    if (threadIdx.x < s) sh[threadIdx.x] += sh[threadIdx.x + s];
    __syncthreads();
  }
  if (threadIdx.x == 0) deq[0] = fmaxf(sh[0] * invn, 1e-5f);
}

__global__ void k_wquant(const float* __restrict__ w, const float* __restrict__ deq,
                         signed char* __restrict__ q, int n) {
  float s = 1.f / deq[0];
  for (int i = blockIdx.x * blockDim.x + threadIdx.x; i < n; i += gridDim.x * blockDim.x) {
    float v = rintf(w[i] * s);
    v = fminf(1.f, fmaxf(-1.f, v));
    q[i] = (signed char)(int)v;
  }
}

__global__ void k_joint_quant(const float* __restrict__ hs, const float* __restrict__ cs,
                              const float* __restrict__ e_h, const float* __restrict__ e_c,
                              signed char* __restrict__ q, float* __restrict__ deq) {
  __shared__ float sh[8];
  int r = blockIdx.x;
  int b = r / SJ_, s = r % SJ_;
  const float* src; const float* e;
  if (s < SCTX_) { src = cs + ((size_t)b * SCTX_ + s) * D_;          e = e_c + (size_t)b * 6 * D_; }
  else           { src = hs + ((size_t)b * SIMG_ + (s - SCTX_)) * D_; e = e_h + (size_t)b * 6 * D_; }
  int tid = threadIdx.x;
  float x[4];
  #pragma unroll
  for (int t = 0; t < 4; ++t) x[t] = src[tid + t * 256];
  float mu = blk_sum256(x[0] + x[1] + x[2] + x[3], sh) * (1.f / D_);
  float vs = 0.f;
  #pragma unroll
  for (int t = 0; t < 4; ++t) { float d = x[t] - mu; vs += d * d; }
  float rstd = rsqrtf(blk_sum256(vs, sh) * (1.f / D_) + EPS_);
  float y[4]; float ss = 0.f, am = 0.f;
  #pragma unroll
  for (int t = 0; t < 4; ++t) {
    int d = tid + t * 256;
    float v = (x[t] - mu) * rstd * (1.f + e[d]) + e[D_ + d];
    y[t] = v; ss += v * v; am = fmaxf(am, fabsf(v));
  }
  float rms  = rsqrtf(blk_sum256(ss, sh) * (1.f / D_) + EPS_);
  float amax = blk_max256(am, sh) * rms;
  float qs = 127.f / fmaxf(amax, 1e-5f);
  #pragma unroll
  for (int t = 0; t < 4; ++t) {
    int d = tid + t * 256;
    float v = rintf(y[t] * rms * qs);
    v = fminf(127.f, fmaxf(-128.f, v));
    q[(size_t)r * D_ + d] = (signed char)(int)v;
  }
  if (tid == 0) deq[r] = 1.f / qs;
}

__launch_bounds__(256)
__global__ void k_gemm_i8(const signed char* __restrict__ Aq, const float* __restrict__ adeq,
                          const signed char* __restrict__ Wq, const float* __restrict__ wdeqp,
                          float* __restrict__ C, int N, int K) {
  __shared__ __align__(16) signed char sA[128 * 80];
  __shared__ __align__(16) signed char sB[64 * 80];
  const int tid  = threadIdx.x;
  const int lane = tid & 31;
  const int wv   = tid >> 5;
  const int wm   = wv & 3, wn = wv >> 2;
  const int m0   = blockIdx.y * 128;
  const int n0   = blockIdx.x * 64;
  const int lrow = lane & 15;
  const int hi   = lane >> 4;
  const int arow = tid >> 2, aseg = tid & 3;

  v8i_t acc[2][2];
  { V8U z; for (int i = 0; i < 8; ++i) z.i[i] = 0;
    acc[0][0] = z.v; acc[0][1] = z.v; acc[1][0] = z.v; acc[1][1] = z.v; }

  for (int k0 = 0; k0 < K; k0 += 64) {
    if (k0) __syncthreads();
    #pragma unroll
    for (int it = 0; it < 2; ++it) {
      int row = arow + it * 64;
      *(v4i_t*)(sA + row * 80 + aseg * 16) =
          *(const v4i_t*)(Aq + (size_t)(m0 + row) * K + k0 + aseg * 16);
    }
    *(v4i_t*)(sB + arow * 80 + aseg * 16) =
        *(const v4i_t*)(Wq + (size_t)(n0 + arow) * K + k0 + aseg * 16);
    if (k0 + 64 < K)
      __builtin_prefetch(Aq + (size_t)(m0 + arow) * K + k0 + 64, 0, 1);
    __syncthreads();

    V8U af[2], bfr[2];
    #pragma unroll
    for (int i = 0; i < 2; ++i) {
      const signed char* ab = sA + (wm * 32 + i * 16 + lrow) * 80 + hi * 8;
      #pragma unroll
      for (int c = 0; c < 4; ++c) af[i].l[c] = *(const long long*)(ab + c * 16);
    }
    #pragma unroll
    for (int j = 0; j < 2; ++j) {
      const signed char* bb = sB + (wn * 32 + j * 16 + lrow) * 80 + hi * 8;
      #pragma unroll
      for (int c = 0; c < 4; ++c) bfr[j].l[c] = *(const long long*)(bb + c * 16);
    }
    #pragma unroll
    for (int i = 0; i < 2; ++i)
      #pragma unroll
      for (int j = 0; j < 2; ++j)
        acc[i][j] = __builtin_amdgcn_wmma_i32_16x16x64_iu8(
            true, af[i].v, true, bfr[j].v, acc[i][j], false, false);
  }

  float wd = wdeqp[0];
  #pragma unroll
  for (int i = 0; i < 2; ++i) {
    #pragma unroll
    for (int j = 0; j < 2; ++j) {
      int col = n0 + wn * 32 + j * 16 + lrow;
      V8U a; a.v = acc[i][j];
      #pragma unroll
      for (int v = 0; v < 8; ++v) {
        int row = m0 + wm * 32 + i * 16 + v + 8 * hi;
        C[(size_t)row * N + col] = (float)a.i[v] * adeq[row] * wd;
      }
    }
  }
}

__global__ void k_ifgate(float* bi, float* bf, int n) {
  for (int i = blockIdx.x * blockDim.x + threadIdx.x; i < n; i += gridDim.x * blockDim.x) {
    float iv = bi[i], fv = bf[i];
    float fs = sigmoidf_(fv);
    bi[i] = siluf_(iv) * (1.f - fs);
    bf[i] = fs;
  }
}

__global__ void k_scan(const float* bi, const float* bf, float* bo) {
  int ch = blockIdx.x * blockDim.x + threadIdx.x;
  int b = ch / D_, d = ch % D_;
  size_t base = (size_t)b * SJ_ * D_ + d;
  float h = 0.f;
  for (int t = 0; t < SJ_; ++t) {
    size_t idx = base + (size_t)t * D_;
    h = bf[idx] * h + bi[idx];
    bo[idx] = h;
  }
}

__global__ void k_gate_norm_quant(const float* __restrict__ o, const float* __restrict__ g,
                                  const float* __restrict__ gn,
                                  signed char* __restrict__ q, float* __restrict__ deq) {
  __shared__ float sh[8];
  int r = blockIdx.x, tid = threadIdx.x;
  const float* orow = o + (size_t)r * D_;
  const float* grow = g + (size_t)r * D_;
  float x[4]; float ss1 = 0.f;
  #pragma unroll
  for (int t = 0; t < 4; ++t) { x[t] = orow[tid + t * 256]; ss1 += x[t] * x[t]; }
  float r1 = rsqrtf(blk_sum256(ss1, sh) * (1.f / D_) + EPS_);
  float y[4]; float ss2 = 0.f, am = 0.f;
  #pragma unroll
  for (int t = 0; t < 4; ++t) {
    int d = tid + t * 256;
    float v = x[t] * r1 * gn[d] * siluf_(grow[d]);
    y[t] = v; ss2 += v * v; am = fmaxf(am, fabsf(v));
  }
  float rms  = rsqrtf(blk_sum256(ss2, sh) * (1.f / D_) + EPS_);
  float amax = blk_max256(am, sh) * rms;
  float qs = 127.f / fmaxf(amax, 1e-5f);
  #pragma unroll
  for (int t = 0; t < 4; ++t) {
    int d = tid + t * 256;
    float v = rintf(y[t] * rms * qs);
    v = fminf(127.f, fmaxf(-128.f, v));
    q[(size_t)r * D_ + d] = (signed char)(int)v;
  }
  if (tid == 0) deq[r] = 1.f / qs;
}

__global__ void k_resmod_quant(const float* __restrict__ src, const float* __restrict__ attn,
                               const float* __restrict__ e, float* __restrict__ resout,
                               signed char* __restrict__ q, float* __restrict__ deq,
                               int S, int attn_off) {
  __shared__ float sh[8];
  int r = blockIdx.x, tid = threadIdx.x;
  int b = r / S, s = r % S;
  const float* srow = src + (size_t)r * D_;
  const float* arow = attn + ((size_t)b * SJ_ + attn_off + s) * D_;
  const float* eb = e + (size_t)b * 6 * D_;
  float h[4];
  #pragma unroll
  for (int t = 0; t < 4; ++t) {
    int d = tid + t * 256;
    float v = srow[d] + eb[2 * D_ + d] * arow[d];
    h[t] = v;
    resout[(size_t)r * D_ + d] = v;
  }
  float mu = blk_sum256(h[0] + h[1] + h[2] + h[3], sh) * (1.f / D_);
  float vs = 0.f;
  #pragma unroll
  for (int t = 0; t < 4; ++t) { float d = h[t] - mu; vs += d * d; }
  float rstd = rsqrtf(blk_sum256(vs, sh) * (1.f / D_) + EPS_);
  float y[4]; float ss = 0.f, am = 0.f;
  #pragma unroll
  for (int t = 0; t < 4; ++t) {
    int d = tid + t * 256;
    float v = (h[t] - mu) * rstd * (1.f + eb[3 * D_ + d]) + eb[4 * D_ + d];
    y[t] = v; ss += v * v; am = fmaxf(am, fabsf(v));
  }
  float rms  = rsqrtf(blk_sum256(ss, sh) * (1.f / D_) + EPS_);
  float amax = blk_max256(am, sh) * rms;
  float qs = 127.f / fmaxf(amax, 1e-5f);
  #pragma unroll
  for (int t = 0; t < 4; ++t) {
    int d = tid + t * 256;
    float v = rintf(y[t] * rms * qs);
    v = fminf(127.f, fmaxf(-128.f, v));
    q[(size_t)r * D_ + d] = (signed char)(int)v;
  }
  if (tid == 0) deq[r] = 1.f / qs;
}

__global__ void k_glu_quant(const float* __restrict__ big,
                            signed char* __restrict__ q, float* __restrict__ deq) {
  __shared__ float sh[8];
  int r = blockIdx.x, tid = threadIdx.x;
  const float* row = big + (size_t)r * 8192;
  float h[16]; float ss = 0.f, am = 0.f;
  #pragma unroll
  for (int t = 0; t < 16; ++t) {
    int n = tid + t * 256;
    float v = siluf_(row[n]) * row[4096 + n];
    h[t] = v; ss += v * v; am = fmaxf(am, fabsf(v));
  }
  float rms  = rsqrtf(blk_sum256(ss, sh) * (1.f / 4096.f) + EPS_);
  float amax = blk_max256(am, sh) * rms;
  float qs = 127.f / fmaxf(amax, 1e-5f);
  #pragma unroll
  for (int t = 0; t < 16; ++t) {
    int n = tid + t * 256;
    float v = rintf(h[t] * rms * qs);
    v = fminf(127.f, fmaxf(-128.f, v));
    q[(size_t)r * 4096 + n] = (signed char)(int)v;
  }
  if (tid == 0) deq[r] = 1.f / qs;
}

__global__ void k_final_add(float* __restrict__ out, const float* __restrict__ mlp,
                            const float* __restrict__ e, int S, int n) {
  for (int i = blockIdx.x * blockDim.x + threadIdx.x; i < n; i += gridDim.x * blockDim.x) {
    int b = i / (S * D_);
    int d = i % D_;
    out[i] += e[(size_t)b * 6 * D_ + 5 * D_ + d] * mlp[i];
  }
}

extern "C" void kernel_launch(void* const* d_in, const int* in_sizes, int n_in,
                              void* d_out, int out_size, void* d_ws, size_t ws_size,
                              hipStream_t stream) {
  (void)in_sizes; (void)n_in; (void)out_size; (void)ws_size;
  const float* hs   = (const float*)d_in[0];
  const float* cs   = (const float*)d_in[1];
  const float* temb = (const float*)d_in[2];
  const float* aw   = (const float*)d_in[3];
  const float* ab   = (const float*)d_in[4];
  const float* acw  = (const float*)d_in[5];
  const float* acb  = (const float*)d_in[6];
  const float* wi   = (const float*)d_in[7];
  const float* wf   = (const float*)d_in[8];
  const float* wg   = (const float*)d_in[9];
  const float* wo   = (const float*)d_in[10];
  const float* gnw  = (const float*)d_in[11];
  const float* ffg  = (const float*)d_in[12];
  const float* ffd  = (const float*)d_in[13];
  const float* fcg  = (const float*)d_in[14];
  const float* fcd  = (const float*)d_in[15];

  float* out_hid = (float*)d_out;
  float* out_ctx = (float*)d_out + (size_t)ROWSH_ * D_;

  char* base = (char*)d_ws;
  size_t off = 0;
  auto alloc = [&](size_t bytes) -> char* {
    char* p = base + off;
    off = (off + bytes + 255) & ~(size_t)255;
    return p;
  };
  float* e_hid = (float*)alloc((size_t)B_ * 6 * D_ * 4);
  float* e_ctx = (float*)alloc((size_t)B_ * 6 * D_ * 4);
  float* wdeq  = (float*)alloc(8 * 4);
  float* part  = (float*)alloc(256 * 4);
  signed char* qwi  = (signed char*)alloc((size_t)D_ * D_);
  signed char* qwf  = (signed char*)alloc((size_t)D_ * D_);
  signed char* qwg  = (signed char*)alloc((size_t)D_ * D_);
  signed char* qwo  = (signed char*)alloc((size_t)D_ * D_);
  signed char* qffg = (signed char*)alloc((size_t)8 * D_ * D_);
  signed char* qffd = (signed char*)alloc((size_t)4 * D_ * D_);
  signed char* qfcg = (signed char*)alloc((size_t)8 * D_ * D_);
  signed char* qfcd = (signed char*)alloc((size_t)4 * D_ * D_);
  signed char* qx   = (signed char*)alloc((size_t)ROWSJ_ * D_);
  float* dqx        = (float*)alloc((size_t)ROWSJ_ * 4);
  signed char* qh   = (signed char*)alloc((size_t)2048 * 4096);
  float* dqh        = (float*)alloc((size_t)2048 * 4);
  float* bi   = (float*)alloc((size_t)ROWSJ_ * D_ * 4);
  float* bf   = (float*)alloc((size_t)ROWSJ_ * D_ * 4);
  float* bg   = (float*)alloc((size_t)ROWSJ_ * D_ * 4);
  float* attn = (float*)alloc((size_t)ROWSJ_ * D_ * 4);
  float* big  = (float*)alloc((size_t)2048 * 8192 * 4);
  float* mlp  = (float*)alloc((size_t)ROWSH_ * D_ * 4);

  k_adaln<<<dim3(48, B_), 128, 0, stream>>>(temb, aw, ab, e_hid);
  k_adaln<<<dim3(48, B_), 128, 0, stream>>>(temb, acw, acb, e_ctx);

  const float* wsrc[8] = { wi, wf, wg, wo, ffg, ffd, fcg, fcd };
  signed char* wq[8]   = { qwi, qwf, qwg, qwo, qffg, qffd, qfcg, qfcd };
  const int    wn[8]   = { D_*D_, D_*D_, D_*D_, D_*D_, 8*D_*D_, 4*D_*D_, 8*D_*D_, 4*D_*D_ };
  for (int m = 0; m < 8; ++m) {
    k_abs_partial<<<256, 256, 0, stream>>>(wsrc[m], wn[m], part);
    k_wscale<<<1, 256, 0, stream>>>(part, 1.f / (float)wn[m], wdeq + m);
    int g = wn[m] / 256 / 8; if (g < 1) g = 1; if (g > 4096) g = 4096;
    k_wquant<<<g, 256, 0, stream>>>(wsrc[m], wdeq + m, wq[m], wn[m]);
  }

  k_joint_quant<<<ROWSJ_, 256, 0, stream>>>(hs, cs, e_hid, e_ctx, qx, dqx);

  k_gemm_i8<<<dim3(16, 72), 256, 0, stream>>>(qx, dqx, qwi, wdeq + 0, bi, D_, D_);
  k_gemm_i8<<<dim3(16, 72), 256, 0, stream>>>(qx, dqx, qwf, wdeq + 1, bf, D_, D_);
  k_gemm_i8<<<dim3(16, 72), 256, 0, stream>>>(qx, dqx, qwg, wdeq + 2, bg, D_, D_);

  k_ifgate<<<4096, 256, 0, stream>>>(bi, bf, ROWSJ_ * D_);
  k_scan<<<(B_ * D_) / 256, 256, 0, stream>>>(bi, bf, bi);

  k_gate_norm_quant<<<ROWSJ_, 256, 0, stream>>>(bi, bg, gnw, qx, dqx);
  k_gemm_i8<<<dim3(16, 72), 256, 0, stream>>>(qx, dqx, qwo, wdeq + 3, attn, D_, D_);

  k_resmod_quant<<<ROWSH_, 256, 0, stream>>>(hs, attn, e_hid, out_hid, qx, dqx, SIMG_, SCTX_);
  for (int c = 0; c < 4; ++c) {
    size_t roff = (size_t)c * 2048;
    k_gemm_i8<<<dim3(128, 16), 256, 0, stream>>>(qx + roff * D_, dqx + roff,
                                                 qffg, wdeq + 4, big, 8 * D_, D_);
    k_glu_quant<<<2048, 256, 0, stream>>>(big, qh, dqh);
    k_gemm_i8<<<dim3(16, 16), 256, 0, stream>>>(qh, dqh, qffd, wdeq + 5,
                                                mlp + roff * D_, D_, 4 * D_);
  }
  k_final_add<<<4096, 256, 0, stream>>>(out_hid, mlp, e_hid, SIMG_, ROWSH_ * D_);

  k_resmod_quant<<<ROWSC_, 256, 0, stream>>>(cs, attn, e_ctx, out_ctx, qx, dqx, SCTX_, 0);
  k_gemm_i8<<<dim3(128, 8), 256, 0, stream>>>(qx, dqx, qfcg, wdeq + 6, big, 8 * D_, D_);
  k_glu_quant<<<ROWSC_, 256, 0, stream>>>(big, qh, dqh);
  k_gemm_i8<<<dim3(16, 8), 256, 0, stream>>>(qh, dqh, qfcd, wdeq + 7, mlp, D_, 4 * D_);
  k_final_add<<<1024, 256, 0, stream>>>(out_ctx, mlp, e_ctx, SCTX_, ROWSC_ * D_);
}
